// NlsqCond_38989713113397
// MI455X (gfx1250) — matrix operand order repaired and checked by reading, not compiled, not run on hardware
//
#include <hip/hip_runtime.h>
#include <hip/hip_bf16.h>
#include <math.h>

// ---------------------------------------------------------------------------
// Types for CDNA5 WMMA (wave32): v16h A/B frags, v8f C/D frag.
// ---------------------------------------------------------------------------
typedef __attribute__((ext_vector_type(16))) _Float16 v16h;
typedef __attribute__((ext_vector_type(8)))  _Float16 v8h;
typedef __attribute__((ext_vector_type(8)))  float    v8f;
typedef __attribute__((ext_vector_type(4)))  float    v4f;

#define LOG_A_CONST 0.39850637f   // log(8*sqrt(3)/9 - 0.05)

// A-matrix 16x32 f16 layout (ISA 7.12.2):
//   lane = hid*16 + m ; element j<8  -> A[m][kb + hid*8 + j]
//                       element j>=8 -> A[m][kb + 16 + hid*8 + (j-8)]
__device__ inline v16h concat8(v8h lo, v8h hi) {
  v16h r;
#pragma unroll
  for (int j = 0; j < 8; ++j) { r[j] = lo[j]; r[j + 8] = hi[j]; }
  return r;
}

// Generic row-major f16 A-frag load (used by cond MLP layer 2 from LDS).
__device__ inline v16h load_a_f16(const _Float16* __restrict__ base, int stride,
                                  int kb, int lane) {
  const int m = lane & 15, hid = lane >> 4;
  const _Float16* p = base + (size_t)m * stride + kb + hid * 8;
  return concat8(*(const v8h*)p, *(const v8h*)(p + 16));
}

// B operand from fp32 row-major BT[n][k] (converted on load).
__device__ inline v16h load_b_f32rowmajor(const float* __restrict__ base, int stride,
                                          int kb, int n0, int lane) {
  const int n = lane & 15, khi = (lane >> 4) * 8;
  const float* p = base + (size_t)(n0 + n) * stride + kb + khi;
  v4f f0 = *(const v4f*)(p + 0);
  v4f f1 = *(const v4f*)(p + 4);
  v4f f2 = *(const v4f*)(p + 16);
  v4f f3 = *(const v4f*)(p + 20);
  v16h r;
#pragma unroll
  for (int j = 0; j < 4; ++j) {
    r[j]      = (_Float16)f0[j];
    r[j + 4]  = (_Float16)f1[j];
    r[j + 8]  = (_Float16)f2[j];
    r[j + 12] = (_Float16)f3[j];
  }
  return r;
}

__device__ inline v16h load_b_lds(const _Float16* __restrict__ row, int koff) {
  return concat8(*(const v8h*)(row + koff), *(const v8h*)(row + koff + 16));
}

__device__ inline v8f wmma_f16(v16h a, v16h b, v8f c) {
  return __builtin_amdgcn_wmma_f32_16x16x32_f16(false, a, false, b,
                                                (short)0, c, false, false);
}

// ---------------------------------------------------------------------------
// Weight repack: (O, Cin, 3) fp32 -> fragment-ordered f16.
// K-order: k = tap*Cin + i.  Fragment order:
//   wpk[ (((mtile*(K/32) + ks)*32 + lane)*16 + j ) ]
//     = W[mtile*16 + (lane&15)][ ks*32 + fragK(lane>>4, j) ]
// so one wave's A-frag load is a fully contiguous 1 KB block (8 cachelines).
// ---------------------------------------------------------------------------
__global__ __launch_bounds__(256) void k_pack(const float* __restrict__ w,
                                              _Float16* __restrict__ wpk, int Cin) {
  const int K = 3 * Cin;
  const size_t total = (size_t)512 * K;
  size_t idx = (size_t)blockIdx.x * 256 + threadIdx.x;
  if (idx >= total) return;
  const int j    = (int)(idx & 15);
  const int lane = (int)((idx >> 4) & 31);
  const int ksteps = K >> 5;
  const int rest  = (int)(idx >> 9);          // mtile*ksteps + ks
  const int ks    = rest % ksteps;
  const int mtile = rest / ksteps;
  const int o   = mtile * 16 + (lane & 15);
  const int hid = lane >> 4;
  const int kk  = ks * 32 + ((j < 8) ? (hid * 8 + j) : (16 + hid * 8 + (j - 8)));
  const int tap = kk / Cin;
  const int i   = kk - tap * Cin;
  wpk[idx] = (_Float16)w[((size_t)o * Cin + i) * 3 + tap];
}

// Fragment-ordered A load: contiguous 32 B per lane.
__device__ inline v16h load_a_packed(const _Float16* __restrict__ wpk,
                                     int mtile, int ksteps, int ksg, int lane) {
  const _Float16* p = wpk + (((size_t)mtile * ksteps + ksg) * 32 + lane) * 16;
  return concat8(*(const v8h*)p, *(const v8h*)(p + 8));
}

// ---------------------------------------------------------------------------
// Cond MLP: c = relu(relu(cond(16) @ w_c1^T + b_c1) @ w_c2^T + b_c2), f16 out.
// One block per 16-batch group, 8 waves; each wave owns 64 output features.
// ---------------------------------------------------------------------------
__global__ __launch_bounds__(256) void k_cond(const float* __restrict__ cond,
                                              const float* __restrict__ w_c1,
                                              const float* __restrict__ b_c1,
                                              const float* __restrict__ w_c2,
                                              const float* __restrict__ b_c2,
                                              _Float16* __restrict__ Cbuf) {
  __shared__ _Float16 c1[16 * 520];          // 16 batch rows x 512 feats (pitch 520)
  const int g = blockIdx.x;
  const int tid = threadIdx.x;
  const int wave = tid >> 5, lane = tid & 31;
  const int m = lane & 15, hid = lane >> 4;

  // ---- layer 1 : M=16 batch, K=16 (zero-padded to 32), N=512 ----
  v16h a1;
  {
    const float* p = cond + (size_t)(g * 16 + m) * 16 + hid * 8;  // cond flat (B,16)
    v4f f0 = *(const v4f*)(p + 0);
    v4f f1 = *(const v4f*)(p + 4);
#pragma unroll
    for (int j = 0; j < 4; ++j) {
      a1[j]      = (_Float16)f0[j];
      a1[j + 4]  = (_Float16)f1[j];
      a1[j + 8]  = (_Float16)0.0f;
      a1[j + 12] = (_Float16)0.0f;
    }
  }
#pragma unroll
  for (int nt = 0; nt < 4; ++nt) {
    const int n0 = wave * 64 + nt * 16;
    v16h bf;
    {
      const float* p = w_c1 + (size_t)(n0 + m) * 16 + hid * 8;   // (N,K) row-major, K=16
      v4f f0 = *(const v4f*)(p + 0);
      v4f f1 = *(const v4f*)(p + 4);
#pragma unroll
      for (int j = 0; j < 4; ++j) {
        bf[j] = (_Float16)f0[j]; bf[j + 4] = (_Float16)f1[j];
        bf[j + 8] = (_Float16)0.0f; bf[j + 12] = (_Float16)0.0f;   // k>=16 zero pad
      }
    }
    v8f c = {};
    c = wmma_f16(a1, bf, c);
    const int col = n0 + m;                 // N index
    const float bias = b_c1[col];
#pragma unroll
    for (int r = 0; r < 8; ++r) {
      const int row = hid * 8 + r;          // M (batch-in-group)
      float v = c[r] + bias;
      v = v > 0.0f ? v : 0.0f;
      c1[row * 520 + col] = (_Float16)v;
    }
  }
  __syncthreads();

  // ---- layer 2 : K=512 over 16 k-steps ----
#pragma unroll
  for (int nt = 0; nt < 4; ++nt) {
    const int n0 = wave * 64 + nt * 16;
    v8f acc = {};
    for (int ks = 0; ks < 16; ++ks) {
      const int kb = ks * 32;
      v16h af = load_a_f16(c1, 520, kb, lane);
      v16h bf = load_b_f32rowmajor(w_c2, 512, kb, n0, lane);
      acc = wmma_f16(af, bf, acc);
    }
    const int col = n0 + m;
    const float bias = b_c2[col];
#pragma unroll
    for (int r = 0; r < 8; ++r) {
      const int row = hid * 8 + r;
      float v = acc[r] + bias;
      v = v > 0.0f ? v : 0.0f;
      Cbuf[(size_t)(g * 16 + row) * 512 + col] = (_Float16)v;
    }
  }
}

// ---------------------------------------------------------------------------
// Embed + assemble conv0 input X0[b][t][ci] (f16), ci<512: embed, ci>=512: cond.
// Only cols 0,1 of w_embed matter (rest multiply structural zeros).
// ---------------------------------------------------------------------------
__global__ __launch_bounds__(256) void k_embed(const float* __restrict__ x,
                                               const float* __restrict__ w_embed,
                                               const float* __restrict__ b_embed,
                                               const _Float16* __restrict__ Cbuf,
                                               _Float16* __restrict__ X0) {
  size_t i = (size_t)blockIdx.x * 256 + threadIdx.x;
  const size_t total = (size_t)256 * 32 * 1024;
  if (i >= total) return;
  const int ci = (int)(i & 1023);
  const size_t bt = i >> 10;
  const int t = (int)(bt & 31);
  const int b = (int)(bt >> 5);
  if (ci < 512) {
    const float x0 = x[((size_t)b * 64 + t) * 2 + 0];
    const float x1 = x[((size_t)b * 64 + t) * 2 + 1];
    const float v = x0 * w_embed[ci * 6 + 0] + x1 * w_embed[ci * 6 + 1] + b_embed[ci];
    X0[i] = (_Float16)v;
  } else {
    X0[i] = Cbuf[(size_t)b * 512 + (ci - 512)];
  }
}

// ---------------------------------------------------------------------------
// Conv1d(k=3,pad=1) + bias + ReLU as WMMA GEMM.  One block per batch element.
// M=512 out channels (32 tiles), N=32 time (2 tiles), K=3*Cin.
// Activations staged in LDS time-major [t+1][i] (zero halo rows 0 and 33),
// in 512-channel chunks; pitch 520 halfs -> conflict-free ds_load_b128.
// Weights read in fragment order: wave A-frag = contiguous 1 KB block.
// ---------------------------------------------------------------------------
__global__ __launch_bounds__(256) void k_conv(const _Float16* __restrict__ Xin,
                                              const _Float16* __restrict__ Wpk,
                                              const float* __restrict__ bias,
                                              _Float16* __restrict__ Xout, int Cin) {
  __shared__ _Float16 Xl[34 * 520];
  const int b = blockIdx.x;
  const int tid = threadIdx.x;
  const int wave = tid >> 5, lane = tid & 31;
  const int K = 3 * Cin;
  const int ksteps = K >> 5;
  const int nchunks = Cin >> 9;             // Cin/512 (1 or 2)

  v8f acc[4][2];
#pragma unroll
  for (int mt = 0; mt < 4; ++mt)
#pragma unroll
    for (int nt = 0; nt < 2; ++nt) { v8f z = {}; acc[mt][nt] = z; }

  for (int ch = 0; ch < nchunks; ++ch) {
    const int cbase = ch << 9;
    __syncthreads();                        // previous chunk fully consumed
    // cooperative fill: 34 rows x 512 halfs, 8-half vectors
    for (int idx = tid; idx < 34 * 64; idx += 256) {
      const int r = idx >> 6;
      const int cc = (idx & 63) * 8;
      v8h v = {};
      if (r >= 1 && r <= 32)
        v = *(const v8h*)(Xin + ((size_t)b * 32 + (r - 1)) * Cin + cbase + cc);
      *(v8h*)(&Xl[r * 520 + cc]) = v;
    }
    __syncthreads();

#pragma unroll
    for (int tap = 0; tap < 3; ++tap) {
      for (int ks = 0; ks < 16; ++ks) {     // 512-channel chunk = 16 k-steps
        const int ib = ks * 32;
        const int ksg = (tap * Cin + cbase + ib) >> 5;   // global k-step
        v16h afr[4];
#pragma unroll
        for (int mt = 0; mt < 4; ++mt) {
          const int mtile = wave + 8 * mt;
          afr[mt] = load_a_packed(Wpk, mtile, ksteps, ksg, lane);
          // prefetch next k-step's fragment block for this mtile
          __builtin_prefetch(Wpk + (((size_t)mtile * ksteps + ksg + 1) * 32 + lane) * 16, 0, 1);
        }
        v16h bfr[2];
#pragma unroll
        for (int nt = 0; nt < 2; ++nt) {
          const int trow = nt * 16 + (lane & 15) + tap;
          bfr[nt] = load_b_lds(&Xl[trow * 520], ib + (lane >> 4) * 8);
        }
#pragma unroll
        for (int mt = 0; mt < 4; ++mt)
#pragma unroll
          for (int nt = 0; nt < 2; ++nt)
            acc[mt][nt] = wmma_f16(afr[mt], bfr[nt], acc[mt][nt]);
      }
    }
  }

  // epilogue: bias + relu + one contiguous 16 B f16 store per tile per lane
  const int n = lane & 15, hid = lane >> 4;
#pragma unroll
  for (int mt = 0; mt < 4; ++mt) {
    const int m0 = (wave + 8 * mt) * 16;
    const int obase = m0 + hid * 8;
    v4f b0 = *(const v4f*)(bias + obase);
    v4f b1 = *(const v4f*)(bias + obase + 4);
#pragma unroll
    for (int nt = 0; nt < 2; ++nt) {
      const int t = nt * 16 + n;
      v8h pk;
#pragma unroll
      for (int r = 0; r < 8; ++r) {
        float v = acc[mt][nt][r] + (r < 4 ? b0[r & 3] : b1[r & 3]);
        v = v > 0.0f ? v : 0.0f;
        pk[r] = (_Float16)v;
      }
      *(v8h*)(Xout + ((size_t)b * 32 + t) * 512 + obase) = pk;
    }
  }
}

// ---------------------------------------------------------------------------
// Output projection (512 -> 10) + flow transform + per-batch logdet reduction.
// One block per batch element; thread = (t, d).
// ---------------------------------------------------------------------------
__global__ __launch_bounds__(64) void k_out(const _Float16* __restrict__ X,
                                            const float* __restrict__ w_out,
                                            const float* __restrict__ b_out,
                                            const float* __restrict__ x,
                                            float* __restrict__ out) {
  __shared__ float red[64];
  const int b = blockIdx.x;
  const int tid = threadIdx.x;
  const int t = tid >> 1, d = tid & 1;

  const _Float16* xr = X + ((size_t)b * 32 + t) * 512;
  float acc[5];
#pragma unroll
  for (int j = 0; j < 5; ++j) acc[j] = b_out[5 * d + j];
  for (int k = 0; k < 512; k += 8) {
    v8h xv = *(const v8h*)(xr + k);
#pragma unroll
    for (int u = 0; u < 8; ++u) {
      const float xf = (float)xv[u];
#pragma unroll
      for (int j = 0; j < 5; ++j)
        acc[j] = fmaf(xf, w_out[(size_t)(5 * d + j) * 512 + k + u], acc[j]);
    }
  }

  const float z2   = x[((size_t)b * 64 + 32 + t) * 2 + d];
  const float a_   = acc[0];
  const float logb = acc[1] * 0.4f;
  const float Bt   = acc[2] * 0.3f;
  const float logd = acc[3] * 0.4f;
  const float f    = acc[4];
  const float b_   = expf(logb);
  const float d_   = expf(logd);
  const float c_   = tanhf(Bt) * expf(LOG_A_CONST + logb - logd);
  const float arg  = d_ * z2 + f;
  const float den  = 1.0f + arg * arg;
  const float z2n  = a_ + b_ * z2 + c_ / den;
  const float term = b_ - 2.0f * c_ * d_ * arg / (den * den);

  out[((size_t)b * 64 + t) * 2 + d]      = x[((size_t)b * 64 + t) * 2 + d]; // z1
  out[((size_t)b * 64 + 32 + t) * 2 + d] = z2n;

  red[tid] = logf(term);
  __syncthreads();
  for (int s = 32; s > 0; s >>= 1) {
    if (tid < s) red[tid] += red[tid + s];
    __syncthreads();
  }
  if (tid == 0) out[(size_t)256 * 64 * 2 + b] = red[0];
}

// ---------------------------------------------------------------------------
// Host-side launcher.
// ---------------------------------------------------------------------------
extern "C" void kernel_launch(void* const* d_in, const int* in_sizes, int n_in,
                              void* d_out, int out_size, void* d_ws, size_t ws_size,
                              hipStream_t stream) {
  (void)in_sizes; (void)n_in; (void)out_size; (void)ws_size;
  const float* x       = (const float*)d_in[0];
  const float* cond    = (const float*)d_in[1];
  const float* w_embed = (const float*)d_in[2];
  const float* b_embed = (const float*)d_in[3];
  const float* w_c1    = (const float*)d_in[4];
  const float* b_c1    = (const float*)d_in[5];
  const float* w_c2    = (const float*)d_in[6];
  const float* b_c2    = (const float*)d_in[7];
  const float* w_conv[5] = {(const float*)d_in[8],  (const float*)d_in[10],
                            (const float*)d_in[12], (const float*)d_in[14],
                            (const float*)d_in[16]};
  const float* b_conv[5] = {(const float*)d_in[9],  (const float*)d_in[11],
                            (const float*)d_in[13], (const float*)d_in[15],
                            (const float*)d_in[17]};
  const float* w_out = (const float*)d_in[18];
  const float* b_out = (const float*)d_in[19];
  float* outp = (float*)d_out;

  char* ws = (char*)d_ws;
  size_t off = 0;
  auto carve = [&](size_t bytes) -> void* {
    void* p = ws + off;
    off = (off + bytes + 255) & ~(size_t)255;
    return p;
  };
  _Float16* Wpk[5];
  Wpk[0] = (_Float16*)carve((size_t)512 * 3072 * 2);
  for (int l = 1; l < 5; ++l) Wpk[l] = (_Float16*)carve((size_t)512 * 1536 * 2);
  _Float16* Cbuf = (_Float16*)carve((size_t)256 * 512 * 2);
  _Float16* X0   = (_Float16*)carve((size_t)256 * 32 * 1024 * 2);
  _Float16* Xa   = (_Float16*)carve((size_t)256 * 32 * 512 * 2);
  _Float16* Xb   = (_Float16*)carve((size_t)256 * 32 * 512 * 2);

  // 1) pack weights to f16, fragment order
  k_pack<<<(512 * 3072 + 255) / 256, 256, 0, stream>>>(w_conv[0], Wpk[0], 1024);
  for (int l = 1; l < 5; ++l)
    k_pack<<<(512 * 1536 + 255) / 256, 256, 0, stream>>>(w_conv[l], Wpk[l], 512);

  // 2) cond MLP (WMMA)
  k_cond<<<16, 256, 0, stream>>>(cond, w_c1, b_c1, w_c2, b_c2, Cbuf);

  // 3) embed + assemble conv0 input
  k_embed<<<(256 * 32 * 1024 + 255) / 256, 256, 0, stream>>>(x, w_embed, b_embed, Cbuf, X0);

  // 4) five conv layers (WMMA GEMMs), ping-pong activations
  k_conv<<<256, 256, 0, stream>>>(X0, Wpk[0], b_conv[0], Xa, 1024);
  k_conv<<<256, 256, 0, stream>>>(Xa, Wpk[1], b_conv[1], Xb, 512);
  k_conv<<<256, 256, 0, stream>>>(Xb, Wpk[2], b_conv[2], Xa, 512);
  k_conv<<<256, 256, 0, stream>>>(Xa, Wpk[3], b_conv[3], Xb, 512);
  k_conv<<<256, 256, 0, stream>>>(Xb, Wpk[4], b_conv[4], Xa, 512);

  // 5) projection + flow + logdet
  k_out<<<256, 64, 0, stream>>>(Xa, w_out, b_out, x, outp);
}
